// MultiHeadAttention_84679575208131
// MI455X (gfx1250) — matrix operand order repaired and checked by reading, not compiled, not run on hardware
//
#include <hip/hip_runtime.h>
#include <cstdint>

// ---------------------------------------------------------------------------
// MHA forward for MI455X (gfx1250, wave32, WMMA f32_16x16x32_f16)
// B=4, S=2048, D=512, H=8, DK=64
// Data movement: async-to-LDS (A tiles) + TDM tensor_load_to_lds (B tiles),
// non-temporal stores for the write-once attn/out tensors.
// ---------------------------------------------------------------------------

#define B_  4
#define S_  2048
#define D_  512
#define H_  8
#define DK_ 64

typedef __attribute__((ext_vector_type(16))) _Float16 v16h;
typedef __attribute__((ext_vector_type(8)))  _Float16 v8h;
typedef __attribute__((ext_vector_type(4)))  _Float16 v4h;
typedef __attribute__((ext_vector_type(8)))  float    v8f;
typedef __attribute__((ext_vector_type(4)))  float    v4f;
typedef __attribute__((ext_vector_type(4)))  int      v4i;
typedef __attribute__((ext_vector_type(8)))  int      v8i;

static __device__ __forceinline__ v8f wmma_f16(v16h a, v16h b, v8f c) {
  return __builtin_amdgcn_wmma_f32_16x16x32_f16(false, a, false, b, (short)0, c,
                                                false, false);
}

// ---- CDNA5 counter waits (builtin if declared, else raw SOPP) -------------
static __device__ __forceinline__ void wait_asynccnt0() {
#if __has_builtin(__builtin_amdgcn_s_wait_asynccnt)
  __builtin_amdgcn_s_wait_asynccnt(0);
#else
  asm volatile("s_wait_asynccnt 0x0" ::: "memory");
#endif
}
static __device__ __forceinline__ void wait_tensorcnt0() {
#if __has_builtin(__builtin_amdgcn_s_wait_tensorcnt)
  __builtin_amdgcn_s_wait_tensorcnt(0);
#else
  asm volatile("s_wait_tensorcnt 0x0" ::: "memory");
#endif
}

// ---- async 16B copy global->LDS (per-lane, ASYNCcnt-tracked) --------------
static __device__ __forceinline__ void async_copy16(uint32_t lds_off,
                                                    const void* g) {
  asm volatile("global_load_async_to_lds_b128 %0, %1, off"
               :: "v"(lds_off), "v"((uint64_t)(uintptr_t)g)
               : "memory");
}

// ---- TDM: one 2-D tile load  global -> LDS  (TENSORcnt-tracked) -----------
// tile_k contiguous elements (dim0) x tile_n rows, f16, row stride ld_elems.
static __device__ __forceinline__ void tdm_load_2d(uint32_t lds_off,
                                                   const void* g,
                                                   int tile_k, int tile_n,
                                                   int ld_elems) {
  const uint64_t ga = (uint64_t)(uintptr_t)g;
  v4i g0;
  g0[0] = 1;                                         // count=1, user mode
  g0[1] = (int)lds_off;                              // lds_addr (bytes)
  g0[2] = (int)(uint32_t)ga;                         // global_addr[31:0]
  g0[3] = (int)((uint32_t)(ga >> 32) | (2u << 30));  // addr[56:32] | type=2
  v8i g1;
  g1[0] = 1 << 16;                                   // mask=0, data_size=2B
  g1[1] = (ld_elems & 0xFFFF) << 16;                 // tensor_dim0[15:0]
  g1[2] = ((ld_elems >> 16) & 0xFFFF) | (int)(0xFFFFu << 16); // dim1[15:0]
  g1[3] = 0 | (tile_k << 16);                        // dim1[31:16]=0, tile_dim0
  g1[4] = tile_n;                                    // tile_dim1, tile_dim2=0
  g1[5] = ld_elems;                                  // dim0_stride[31:0]
  g1[6] = 0;                                         // dim0_stride[47:32]
  g1[7] = 0;
  const v4i gz = {0, 0, 0, 0};
  asm volatile("tensor_load_to_lds %0, %1, %2, %3"
               :: "s"(g0), "s"(g1), "s"(gz), "s"(gz)
               : "memory");
}

// ---- WMMA fragment builders (layouts per CDNA5 ISA) ------------
static __device__ __forceinline__ v16h
load_a_f16(const _Float16* __restrict__ X, int ld, int row0, int k0, int lane) {
  const int hl = lane >> 4, m = lane & 15;
  const _Float16* p = X + (size_t)(row0 + m) * ld + (k0 + hl * 8);
  v8h lo = *(const v8h*)p;
  v8h hi = *(const v8h*)(p + 16);
  v16h a;
#pragma unroll
  for (int i = 0; i < 8; ++i) { a[i] = lo[i]; a[i + 8] = hi[i]; }
  return a;
}

static __device__ __forceinline__ v16h
load_b_f16(const _Float16* __restrict__ W, int ld, int n0, int k0, int lane) {
  const int kh = lane >> 4, n = lane & 15;
  const _Float16* p = W + (size_t)(n0 + n) * ld + (k0 + kh * 8);
  v8h lo = *(const v8h*)p;
  v8h hi = *(const v8h*)(p + 16);
  v16h b;
#pragma unroll
  for (int i = 0; i < 8; ++i) { b[i] = lo[i]; b[i + 8] = hi[i]; }
  return b;
}

static __device__ __forceinline__ v16h
load_a_lds_f32(const float* __restrict__ sc, int ld, int k0, int lane) {
  const int hl = lane >> 4, m = lane & 15;
  const float* p = sc + (size_t)m * ld + (k0 + hl * 8);
  v16h a;
#pragma unroll
  for (int i = 0; i < 8; ++i) {
    a[i]     = (_Float16)p[i];
    a[i + 8] = (_Float16)p[i + 16];
  }
  return a;
}

// ---------------------------------------------------------------------------
// f32 -> f16 conversion (one-shot streaming read -> NT load)
// ---------------------------------------------------------------------------
__global__ __launch_bounds__(256) void cvt_f32_f16(const float* __restrict__ in,
                                                   _Float16* __restrict__ out,
                                                   int n4) {
  int i = blockIdx.x * 256 + threadIdx.x;
  if (i < n4) {
    v4f v = __builtin_nontemporal_load((const v4f*)in + i);
    v4h o;
    o[0] = (_Float16)v[0]; o[1] = (_Float16)v[1];
    o[2] = (_Float16)v[2]; o[3] = (_Float16)v[3];
    ((v4h*)out)[i] = o;
  }
}

// ---------------------------------------------------------------------------
// GEMM: Y = X @ W^T + bias.  X: [M x K] f16 row-major, W: [N x K] f16.
// mode 0: Y f16 [M x N];  mode 1: Y f16 V^T-per-head;  mode 2: Y f32 [M x N]
// Block 256 thr = 8 waves (4x2), wave tile 32x32 -> 128x64 block, K-step 32.
// A tile (128x32) staged via async-to-LDS; B tile (64x32) via one TDM
// descriptor issued by wave 7.  Double-buffered.
// ---------------------------------------------------------------------------
__global__ __launch_bounds__(256) void gemm_xwT(const _Float16* __restrict__ X,
                                                const _Float16* __restrict__ W,
                                                const float* __restrict__ bias,
                                                void* __restrict__ Y,
                                                int M, int N, int K, int mode) {
  __shared__ _Float16 As[2][128 * 32];   // 16 KB
  __shared__ _Float16 Bs[2][64 * 32];    //  8 KB
  const int tid = threadIdx.x;
  const int lane = tid & 31;
  const int wave = tid >> 5;
  const int wm = wave >> 1, wn = wave & 1;
  const int rowBlk = blockIdx.x * 128;
  const int colBlk = blockIdx.y * 64;
  const uint32_t aBase = (uint32_t)(uintptr_t)&As[0][0];
  const uint32_t bBase = (uint32_t)(uintptr_t)&Bs[0][0];

  auto stage = [&](int buf, int k0) {
    // A: 8 KB = 512 x 16B chunks, 2 per thread
#pragma unroll
    for (int i = 0; i < 2; ++i) {
      const int chunk = tid + 256 * i;          // 0..511
      const int r = chunk >> 2, piece = chunk & 3;
      async_copy16(aBase + (uint32_t)(buf * 8192 + r * 64 + piece * 16),
                   X + (size_t)(rowBlk + r) * K + k0 + piece * 8);
    }
    // B: one TDM 2-D tile (32 contiguous K elems x 64 rows)
    if (wave == 7)
      tdm_load_2d(bBase + (uint32_t)(buf * 4096),
                  W + (size_t)colBlk * K + k0, 32, 64, K);
  };
  auto settle = [&]() {
    wait_asynccnt0();
    if (wave == 7) wait_tensorcnt0();
    __syncthreads();
  };

  stage(0, 0);
  settle();

  v8f acc[2][2] = {};
  for (int k0 = 0; k0 < K; k0 += 32) {
    const int cur = (k0 >> 5) & 1;
    if (k0 + 32 < K) stage(cur ^ 1, k0 + 32);

    v16h a0 = load_a_f16(&As[cur][0], 32, wm * 32, 0, lane);
    v16h a1 = load_a_f16(&As[cur][0], 32, wm * 32 + 16, 0, lane);
    v16h b0 = load_b_f16(&Bs[cur][0], 32, wn * 32, 0, lane);
    v16h b1 = load_b_f16(&Bs[cur][0], 32, wn * 32 + 16, 0, lane);
    acc[0][0] = wmma_f16(a0, b0, acc[0][0]);
    acc[0][1] = wmma_f16(a0, b1, acc[0][1]);
    acc[1][0] = wmma_f16(a1, b0, acc[1][0]);
    acc[1][1] = wmma_f16(a1, b1, acc[1][1]);

    settle();   // producers of next buffer done; safe to flip
  }

  const int hl = lane >> 4, nloc = lane & 15;
#pragma unroll
  for (int i = 0; i < 2; ++i) {
#pragma unroll
    for (int j = 0; j < 2; ++j) {
      const int n = colBlk + wn * 32 + j * 16 + nloc;
      const float bv = bias[n];
#pragma unroll
      for (int r = 0; r < 8; ++r) {
        const int m = rowBlk + wm * 32 + i * 16 + r + 8 * hl;
        const float v = acc[i][j][r] + bv;
        if (mode == 2) {
          __builtin_nontemporal_store(v, (float*)Y + (size_t)m * N + n);
        } else if (mode == 0) {
          ((_Float16*)Y)[(size_t)m * N + n] = (_Float16)v;
        } else {  // mode 1: V^T per-head store
          const int b = m >> 11, s = m & (S_ - 1);   // S = 2048
          const int h = n >> 6, d = n & (DK_ - 1);   // DK = 64
          ((_Float16*)Y)[((size_t)((b * H_ + h) * DK_ + d) << 11) + s] =
              (_Float16)v;
        }
      }
    }
  }
}

// ---------------------------------------------------------------------------
// Fused attention: 16 query rows of one (b,h) per workgroup.
//   phase 1: scores(16x2048) = Q K^T / 8 via WMMA -> LDS (f32)
//   phase 2: row softmax in LDS; stream probs to d_out.attn with NT stores
//   phase 3: ctx(16x64) = P @ V via WMMA (P from LDS, V^T contiguous)
// Block 128 thr (4 waves).  Dyn LDS: 16*2048 f32 + 512 f32 = 133120 B.
// ---------------------------------------------------------------------------
__global__ __launch_bounds__(128) void attn_fused(
    const _Float16* __restrict__ Q,    // [B*S, D] f16
    const _Float16* __restrict__ Km,   // [B*S, D] f16
    const _Float16* __restrict__ Vt,   // [B*H*DK, S] f16
    float* __restrict__ attn,          // [B*H, S, S] f32
    _Float16* __restrict__ ctx) {      // [B*S, D] f16
  const int s0 = blockIdx.x * 16;
  const int bh = blockIdx.y;
  const int b = bh >> 3, h = bh & 7;
  const int lane = threadIdx.x & 31;
  const int wave = threadIdx.x >> 5;

  extern __shared__ float sc[];        // [16][2048]
  float* red = sc + 16 * S_;           // 512 floats reduction scratch

  const _Float16* qbase = Q + (size_t)(b * S_ + s0) * D_ + h * DK_;
  const v16h aq0 = load_a_f16(qbase, D_, 0, 0, lane);
  const v16h aq1 = load_a_f16(qbase, D_, 0, 32, lane);

  const int hl = lane >> 4, nloc = lane & 15;

  // ---- phase 1: scores -> LDS ----
  const _Float16* kbase = Km + (size_t)(b * S_) * D_ + h * DK_;
  for (int t = wave; t < S_ / 16; t += 4) {
    const int j0 = t * 16;
    v16h b0 = load_b_f16(kbase, D_, j0, 0, lane);
    v16h b1 = load_b_f16(kbase, D_, j0, 32, lane);
    v8f s = {};
    s = wmma_f16(aq0, b0, s);
    s = wmma_f16(aq1, b1, s);
#pragma unroll
    for (int r = 0; r < 8; ++r)
      sc[(size_t)(r + 8 * hl) * S_ + j0 + nloc] = s[r] * 0.125f;  // 1/sqrt(DK)
  }
  __syncthreads();

  // ---- phase 2: softmax ----
  const int row = threadIdx.x >> 3;
  const int sub = threadIdx.x & 7;
  float mx = -3.402823466e38f;
  for (int k = 0; k < S_ / 8; ++k)
    mx = fmaxf(mx, sc[(size_t)row * S_ + sub + k * 8]);
  red[threadIdx.x] = mx;
  __syncthreads();
  if (threadIdx.x < 16) {
    float m2 = red[threadIdx.x * 8];
#pragma unroll
    for (int i = 1; i < 8; ++i) m2 = fmaxf(m2, red[threadIdx.x * 8 + i]);
    red[256 + threadIdx.x] = m2;
  }
  __syncthreads();
  const float rmax = red[256 + row];
  float sum = 0.f;
  for (int k = 0; k < S_ / 8; ++k) {
    const int idx = row * S_ + sub + k * 8;
    const float e = __expf(sc[idx] - rmax);
    sc[idx] = e;
    sum += e;
  }
  red[128 + threadIdx.x] = sum;
  __syncthreads();
  if (threadIdx.x < 16) {
    float s2 = 0.f;
#pragma unroll
    for (int i = 0; i < 8; ++i) s2 += red[128 + threadIdx.x * 8 + i];
    red[272 + threadIdx.x] = 1.0f / s2;
  }
  __syncthreads();

  // normalize in LDS + NT-stream attn rows (written once, never re-read)
  float* attn_row = attn + ((size_t)bh * S_ + s0) * S_;
  for (int idx = threadIdx.x; idx < 16 * S_; idx += 128) {
    const int r = idx >> 11, c = idx & (S_ - 1);
    const float v = sc[idx] * red[272 + r];
    sc[idx] = v;
    __builtin_nontemporal_store(v, attn_row + (size_t)r * S_ + c);
  }
  __syncthreads();

  // ---- phase 3: ctx = P @ V ----
  const _Float16* vbase = Vt + (size_t)bh * DK_ * S_;  // [DK rows][S cols]
  v8f c0 = {}, c1 = {};
  for (int j0 = 0; j0 < S_; j0 += 64) {
    v16h pa0 = load_a_lds_f32(sc, S_, j0, lane);
    v16h pa1 = load_a_lds_f32(sc, S_, j0 + 32, lane);
    v16h vb0 = load_b_f16(vbase, S_, wave * 16, j0, lane);
    v16h vb1 = load_b_f16(vbase, S_, wave * 16, j0 + 32, lane);
    c0 = wmma_f16(pa0, vb0, c0);
    c1 = wmma_f16(pa1, vb1, c1);
  }
  const v8f cs = c0 + c1;
  _Float16* cbase =
      ctx + (size_t)(b * S_ + s0) * D_ + h * DK_ + wave * 16 + nloc;
#pragma unroll
  for (int r = 0; r < 8; ++r)
    cbase[(size_t)(r + 8 * hl) * D_] = (_Float16)cs[r];
}

// ---------------------------------------------------------------------------
// Host-side orchestration
// ---------------------------------------------------------------------------
extern "C" void kernel_launch(void* const* d_in, const int* in_sizes, int n_in,
                              void* d_out, int out_size, void* d_ws,
                              size_t ws_size, hipStream_t stream) {
  const float* query = (const float*)d_in[0];
  const float* key   = (const float*)d_in[1];
  const float* value = (const float*)d_in[2];
  const float* wq_w  = (const float*)d_in[3];
  const float* wq_b  = (const float*)d_in[4];
  const float* wk_w  = (const float*)d_in[5];
  const float* wk_b  = (const float*)d_in[6];
  const float* wv_w  = (const float*)d_in[7];
  const float* wv_b  = (const float*)d_in[8];
  const float* wo_w  = (const float*)d_in[9];
  const float* wo_b  = (const float*)d_in[10];

  const size_t nAct = (size_t)B_ * S_ * D_;   // 4,194,304
  const size_t nW   = (size_t)D_ * D_;        // 262,144

  char* ws = (char*)d_ws;
  size_t off = 0;
  auto take = [&](size_t bytes) { char* p = ws + off; off += bytes; return p; };
  _Float16* xq16 = (_Float16*)take(nAct * 2);
  _Float16* xk16 = (_Float16*)take(nAct * 2);
  _Float16* xv16 = (_Float16*)take(nAct * 2);
  _Float16* q16  = (_Float16*)take(nAct * 2);
  _Float16* k16  = (_Float16*)take(nAct * 2);
  _Float16* vt16 = (_Float16*)take(nAct * 2);
  _Float16* c16  = (_Float16*)take(nAct * 2);
  _Float16* wq16 = (_Float16*)take(nW * 2);
  _Float16* wk16 = (_Float16*)take(nW * 2);
  _Float16* wv16 = (_Float16*)take(nW * 2);
  _Float16* wo16 = (_Float16*)take(nW * 2);

  // 1) precision conversion
  {
    const int a4 = (int)(nAct / 4), w4 = (int)(nW / 4);
    cvt_f32_f16<<<(a4 + 255) / 256, 256, 0, stream>>>(query, xq16, a4);
    cvt_f32_f16<<<(a4 + 255) / 256, 256, 0, stream>>>(key,   xk16, a4);
    cvt_f32_f16<<<(a4 + 255) / 256, 256, 0, stream>>>(value, xv16, a4);
    cvt_f32_f16<<<(w4 + 255) / 256, 256, 0, stream>>>(wq_w, wq16, w4);
    cvt_f32_f16<<<(w4 + 255) / 256, 256, 0, stream>>>(wk_w, wk16, w4);
    cvt_f32_f16<<<(w4 + 255) / 256, 256, 0, stream>>>(wv_w, wv16, w4);
    cvt_f32_f16<<<(w4 + 255) / 256, 256, 0, stream>>>(wo_w, wo16, w4);
  }

  // 2) QKV projections (M=8192, N=512, K=512)
  const int M = B_ * S_;
  dim3 ggrid(M / 128, D_ / 64);
  gemm_xwT<<<ggrid, 256, 0, stream>>>(xq16, wq16, wq_b, q16,  M, D_, D_, 0);
  gemm_xwT<<<ggrid, 256, 0, stream>>>(xk16, wk16, wk_b, k16,  M, D_, D_, 0);
  gemm_xwT<<<ggrid, 256, 0, stream>>>(xv16, wv16, wv_b, vt16, M, D_, D_, 1);

  // 3) fused scores -> softmax -> attn-out -> ctx
  float* attn = (float*)d_out + nAct;           // d_out = [out | attn]
  const size_t ldsBytes = (size_t)(16 * S_ + 512) * sizeof(float);  // 133,120
  attn_fused<<<dim3(S_ / 16, B_ * H_), 128, ldsBytes, stream>>>(
      q16, k16, vt16, attn, c16);

  // 4) output projection -> f32 d_out
  gemm_xwT<<<ggrid, 256, 0, stream>>>(c16, wo16, wo_b, (float*)d_out,
                                      M, D_, D_, 2);
}
